// EfficientMultiHeadLocalSelfAttention_70531952934971
// MI455X (gfx1250) — matrix operand order repaired and checked by reading, not compiled, not run on hardware
//
#include <hip/hip_runtime.h>

#define B_  2
#define T_  8192
#define D_  512
#define H_  8
#define W_  256
#define DH_ 64
#define NC_ 32

typedef __attribute__((ext_vector_type(16))) __bf16 v16bf;
typedef __attribute__((ext_vector_type(8)))  float  v8f;
typedef __attribute__((ext_vector_type(4)))  unsigned int v4u;
typedef unsigned short bf16u;

// float -> bf16 (round-to-nearest-even), bit-level
__device__ __forceinline__ bf16u f2bf(float f) {
  unsigned int u = __float_as_uint(f);
  u += 0x7FFFu + ((u >> 16) & 1u);
  return (bf16u)(u >> 16);
}

// bf16 {p..p+7, p+16..p+23} -> v16bf  (B-matrix fragment: lane=col, K table order)
__device__ __forceinline__ v16bf load16(const bf16u* p) {
  union { v16bf v; v4u u[2]; } t;
  t.u[0] = *(const v4u*)(p);
  t.u[1] = *(const v4u*)(p + 16);
  return t.v;
}
// A-matrix fragment: elements {kb..kb+7, kb+16..kb+23} of one row
__device__ __forceinline__ v16bf loadA(const bf16u* row, int kb) {
  union { v16bf v; v4u u[2]; } t;
  t.u[0] = *(const v4u*)(row + kb);
  t.u[1] = *(const v4u*)(row + kb + 16);
  return t.v;
}
__device__ __forceinline__ v8f wmma_bf16(v16bf a, v16bf b, v8f c) {
  return __builtin_amdgcn_wmma_f32_16x16x32_bf16(false, a, false, b, (short)0, c, false, false);
}

// CDNA5 async copy global -> LDS (16B per lane), tracked by ASYNCcnt
__device__ __forceinline__ void async_ld_b128(const bf16u* gsrc, bf16u* ldst) {
  unsigned l = (unsigned)(uintptr_t)ldst;                 // flat LDS addr low 32b = LDS offset
  unsigned long long g = (unsigned long long)(uintptr_t)gsrc;
  asm volatile("global_load_async_to_lds_b128 %0, %1, off" :: "v"(l), "v"(g) : "memory");
}

// ---------------------------------------------------------------- convert
__global__ void cvt_f32_bf16(const float* __restrict__ s, bf16u* __restrict__ d, int n) {
  int i = blockIdx.x * 256 + threadIdx.x;
  if (i < n) d[i] = f2bf(s[i]);
}

// ---------------------------------------------------------------- fused QKV projection
__launch_bounds__(256, 2)
__global__ void qkv_proj(const bf16u* __restrict__ X,
                         const bf16u* __restrict__ Wqb, const bf16u* __restrict__ Wkb,
                         const bf16u* __restrict__ Wvb,
                         const float* __restrict__ bq, const float* __restrict__ bk,
                         const float* __restrict__ bv,
                         bf16u* __restrict__ Q, bf16u* __restrict__ K, bf16u* __restrict__ V)
{
  const float QSCALE = 0.125f * 1.4426950408889634f; // d^-1/2 * log2(e), folded into Q
  int lane = threadIdx.x & 31;
  int l15  = lane & 15;
  int task = blockIdx.x * 8 + (threadIdx.x >> 5);
  int mt = task >> 3;       // 1024 m-tiles of 16 rows
  int ns = task & 7;        // 8 strips of 64 cols (== head)
  int m0 = mt * 16;
  const bf16u* arow = X + (size_t)(m0 + l15) * D_;
  int akb = (lane >= 16) ? 8 : 0;
  int bks = (lane >= 16) ? 8 : 0;
  v8f aq[4] = {}, ak[4] = {}, av[4] = {};
  for (int kt = 0; kt < 16; ++kt) {
    v16bf a = loadA(arow, kt * 32 + akb);
#pragma unroll
    for (int nt = 0; nt < 4; ++nt) {
      size_t wrow = (size_t)(ns * 64 + nt * 16 + l15) * D_ + kt * 32 + bks;
      aq[nt] = wmma_bf16(a, load16(Wqb + wrow), aq[nt]);
      ak[nt] = wmma_bf16(a, load16(Wkb + wrow), ak[nt]);
      av[nt] = wmma_bf16(a, load16(Wvb + wrow), av[nt]);
    }
  }
  int mh = (lane >= 16) ? 8 : 0;
  int b = m0 / T_;
  int t = m0 % T_;
#pragma unroll
  for (int nt = 0; nt < 4; ++nt) {
    int n  = ns * 64 + nt * 16 + l15;
    int dd = nt * 16 + l15;
    float vq = bq[n], vk = bk[n], vv = bv[n];
    size_t base = ((size_t)(b * H_ + ns) * T_ + t + mh) * DH_ + dd;
#pragma unroll
    for (int r = 0; r < 8; ++r) {
      size_t idx = base + (size_t)r * DH_;
      Q[idx] = f2bf((aq[nt][r] + vq) * QSCALE);
      K[idx] = f2bf(ak[nt][r] + vk);
      V[idx] = f2bf(av[nt][r] + vv);
    }
  }
}

// ---------------------------------------------------------------- local windowed attention
// one block per (b,h,chunk); 16 waves, each owns a 16-row query tile.
// window streamed in 64-key blocks (invalid blocks 64-aligned -> skipped, no masking).
// K tiles: async global->LDS prefetch, double buffered (ASYNCcnt).
// V tiles: register prefetch + transpose-store into double-buffered LDS.
__launch_bounds__(512, 1)
__global__ void local_attn(const bf16u* __restrict__ Q, const bf16u* __restrict__ K,
                           const bf16u* __restrict__ V, bf16u* __restrict__ AO)
{
  __shared__ __align__(16) bf16u Kt[2][64][64];    // [buf][key][d]
  __shared__ __align__(16) bf16u Vt[2][64][72];    // [buf][d][key] (padded)
  __shared__ __align__(16) bf16u Ps[16][16 * 64];  // per-wave P scratch (16x64)

  int c = blockIdx.x & (NC_ - 1);
  int h = (blockIdx.x >> 5) & (H_ - 1);
  int b = blockIdx.x >> 8;
  int bh = b * H_ + h;
  int tid = threadIdx.x;
  int lane = tid & 31;
  int wave = tid >> 5;
  int l15 = lane & 15;
  int hi = (lane >= 16) ? 1 : 0;

  // Q fragments live in registers for the whole window
  const bf16u* qrow = Q + ((size_t)bh * T_ + c * W_ + wave * 16 + l15) * DH_;
  v16bf qa0 = loadA(qrow, hi * 8);
  v16bf qa1 = loadA(qrow, 32 + hi * 8);

  v8f o[4] = {};
  float rmaxv[8], rsumv[8];
#pragma unroll
  for (int r = 0; r < 8; ++r) { rmaxv[r] = -3.0e38f; rsumv[r] = 0.f; }

  int blk0 = (c == 0) ? 4 : 0;
  int blk1 = (c == NC_ - 1) ? 8 : 12;

  // per-thread copy roles
  int keyl = tid >> 3, part = tid & 7;              // K: 16B chunk per thread
  int vkey = tid & 63, vdd0 = (tid >> 6) * 8;       // V: 8 elems of one key-row
  const bf16u* kbase = K + (size_t)bh * T_ * DH_;
  const bf16u* vbase = V + (size_t)bh * T_ * DH_;

  // prologue: start block blk0 in flight
  v4u vreg;
  {
    int kabs = c * W_ - W_ + blk0 * 64;
    async_ld_b128(kbase + (size_t)(kabs + keyl) * DH_ + part * 8,
                  &Kt[blk0 & 1][keyl][part * 8]);
    vreg = *(const v4u*)(vbase + (size_t)(kabs + vkey) * DH_ + vdd0);
  }

  for (int blk = blk0; blk < blk1; ++blk) {
    int cur = blk & 1;
    // commit prefetched V tile into LDS (transposed)
    {
      union { v4u u; bf16u e[8]; } tv;
      tv.u = vreg;
#pragma unroll
      for (int i = 0; i < 8; ++i) Vt[cur][vdd0 + i][vkey] = tv.e[i];
    }
    // kick off next block's K (async) and V (register) prefetch
    bool hn = (blk + 1 < blk1);
    if (hn) {
      int kabs = c * W_ - W_ + (blk + 1) * 64;
      async_ld_b128(kbase + (size_t)(kabs + keyl) * DH_ + part * 8,
                    &Kt[1 - cur][keyl][part * 8]);
      vreg = *(const v4u*)(vbase + (size_t)(kabs + vkey) * DH_ + vdd0);
      asm volatile("s_wait_asynccnt 0x1" ::: "memory");   // current K tile landed
    } else {
      asm volatile("s_wait_asynccnt 0x0" ::: "memory");
    }
    __syncthreads();

    // S = Q * K^T  (16 x 64 per wave)
    v8f s[4];
#pragma unroll
    for (int nt = 0; nt < 4; ++nt) {
      const bf16u* krow = &Kt[cur][nt * 16 + l15][0];
      v8f acc = {};
      acc = wmma_bf16(qa0, load16(krow + hi * 8), acc);
      acc = wmma_bf16(qa1, load16(krow + 32 + hi * 8), acc);
      s[nt] = acc;
    }

    // online softmax update (rows live across 16-lane halves)
    float scl[8];
#pragma unroll
    for (int r = 0; r < 8; ++r) {
      float mx = fmaxf(fmaxf(s[0][r], s[1][r]), fmaxf(s[2][r], s[3][r]));
      mx = fmaxf(mx, __shfl_xor(mx, 1, 32));
      mx = fmaxf(mx, __shfl_xor(mx, 2, 32));
      mx = fmaxf(mx, __shfl_xor(mx, 4, 32));
      mx = fmaxf(mx, __shfl_xor(mx, 8, 32));
      float nm = fmaxf(rmaxv[r], mx);
      float sc = exp2f(rmaxv[r] - nm);
      rmaxv[r] = nm;
      float rs = 0.f;
#pragma unroll
      for (int nt = 0; nt < 4; ++nt) {
        float p = exp2f(s[nt][r] - nm);
        s[nt][r] = p;
        rs += p;
      }
      rs += __shfl_xor(rs, 1, 32);
      rs += __shfl_xor(rs, 2, 32);
      rs += __shfl_xor(rs, 4, 32);
      rs += __shfl_xor(rs, 8, 32);
      rsumv[r] = rsumv[r] * sc + rs;
      scl[r] = sc;
    }
#pragma unroll
    for (int nt = 0; nt < 4; ++nt)
#pragma unroll
      for (int r = 0; r < 8; ++r) o[nt][r] *= scl[r];

    // P: C-layout -> per-wave LDS scratch -> A-layout
    bf16u* ps = &Ps[wave][0];
    int mh = hi * 8;
#pragma unroll
    for (int nt = 0; nt < 4; ++nt)
#pragma unroll
      for (int r = 0; r < 8; ++r)
        ps[(r + mh) * 64 + nt * 16 + l15] = f2bf(s[nt][r]);

    v16bf pa0 = loadA(ps + l15 * 64, hi * 8);
    v16bf pa1 = loadA(ps + l15 * 64, 32 + hi * 8);
#pragma unroll
    for (int nt = 0; nt < 4; ++nt) {
      const bf16u* vrow = &Vt[cur][nt * 16 + l15][0];
      o[nt] = wmma_bf16(pa0, load16(vrow + hi * 8), o[nt]);
      o[nt] = wmma_bf16(pa1, load16(vrow + 32 + hi * 8), o[nt]);
    }
    __syncthreads();
  }

  // normalize, store to AO in [B, T, D] (ready for output projection)
  int mh = hi * 8;
  int t0 = c * W_ + wave * 16;
#pragma unroll
  for (int r = 0; r < 8; ++r) {
    float inv = 1.0f / rsumv[r];
    size_t row = (size_t)b * T_ + t0 + r + mh;
#pragma unroll
    for (int nt = 0; nt < 4; ++nt)
      AO[row * D_ + h * DH_ + nt * 16 + l15] = f2bf(o[nt][r] * inv);
  }
}

// ---------------------------------------------------------------- output projection
__launch_bounds__(256, 2)
__global__ void out_proj(const bf16u* __restrict__ AO, const bf16u* __restrict__ Wob,
                         const float* __restrict__ bo, float* __restrict__ out)
{
  int lane = threadIdx.x & 31;
  int l15 = lane & 15;
  int task = blockIdx.x * 8 + (threadIdx.x >> 5);
  int mt = task >> 3;
  int ns = task & 7;
  int m0 = mt * 16;
  const bf16u* arow = AO + (size_t)(m0 + l15) * D_;
  int akb = (lane >= 16) ? 8 : 0;
  int bks = (lane >= 16) ? 8 : 0;
  v8f acc[4] = {};
  for (int kt = 0; kt < 16; ++kt) {
    v16bf a = loadA(arow, kt * 32 + akb);
#pragma unroll
    for (int nt = 0; nt < 4; ++nt) {
      const bf16u* wr = Wob + (size_t)(ns * 64 + nt * 16 + l15) * D_ + kt * 32 + bks;
      acc[nt] = wmma_bf16(a, load16(wr), acc[nt]);
    }
  }
  int mh = (lane >= 16) ? 8 : 0;
#pragma unroll
  for (int nt = 0; nt < 4; ++nt) {
    int n = ns * 64 + nt * 16 + l15;
    float bias = bo[n];
#pragma unroll
    for (int r = 0; r < 8; ++r)
      out[(size_t)(m0 + mh + r) * D_ + n] = acc[nt][r] + bias;
  }
}

// ---------------------------------------------------------------- launch
extern "C" void kernel_launch(void* const* d_in, const int* in_sizes, int n_in,
                              void* d_out, int out_size, void* d_ws, size_t ws_size,
                              hipStream_t stream) {
  (void)in_sizes; (void)n_in; (void)out_size; (void)ws_size;
  const float* x  = (const float*)d_in[0];
  const float* Wq = (const float*)d_in[1];
  const float* bq = (const float*)d_in[2];
  const float* Wk = (const float*)d_in[3];
  const float* bk = (const float*)d_in[4];
  const float* Wv = (const float*)d_in[5];
  const float* bv = (const float*)d_in[6];
  const float* Wo = (const float*)d_in[7];
  const float* bo = (const float*)d_in[8];
  float* out = (float*)d_out;

  const size_t nx = (size_t)B_ * T_ * D_;   // 8,388,608
  const size_t nw = (size_t)D_ * D_;        // 262,144
  char* ws = (char*)d_ws;
  bf16u* xb  = (bf16u*)ws; ws += nx * 2;
  bf16u* wqb = (bf16u*)ws; ws += nw * 2;
  bf16u* wkb = (bf16u*)ws; ws += nw * 2;
  bf16u* wvb = (bf16u*)ws; ws += nw * 2;
  bf16u* wob = (bf16u*)ws; ws += nw * 2;
  bf16u* Qb  = (bf16u*)ws; ws += nx * 2;
  bf16u* Kb  = (bf16u*)ws; ws += nx * 2;
  bf16u* Vb  = (bf16u*)ws; ws += nx * 2;
  bf16u* AO  = (bf16u*)ws; ws += nx * 2;

  cvt_f32_bf16<<<(int)((nx + 255) / 256), 256, 0, stream>>>(x,  xb,  (int)nx);
  cvt_f32_bf16<<<(int)((nw + 255) / 256), 256, 0, stream>>>(Wq, wqb, (int)nw);
  cvt_f32_bf16<<<(int)((nw + 255) / 256), 256, 0, stream>>>(Wk, wkb, (int)nw);
  cvt_f32_bf16<<<(int)((nw + 255) / 256), 256, 0, stream>>>(Wv, wvb, (int)nw);
  cvt_f32_bf16<<<(int)((nw + 255) / 256), 256, 0, stream>>>(Wo, wob, (int)nw);

  qkv_proj<<<1024, 256, 0, stream>>>(xb, wqb, wkb, wvb, bq, bk, bv, Qb, Kb, Vb);
  local_attn<<<B_ * H_ * NC_, 512, 0, stream>>>(Qb, Kb, Vb, AO);
  out_proj<<<1024, 256, 0, stream>>>(AO, wob, bo, out);
}
